// RGCNpre_55405078118531
// MI455X (gfx1250) — matrix operand order repaired and checked by reading, not compiled, not run on hardware
//
#include <hip/hip_runtime.h>

typedef __attribute__((ext_vector_type(16))) _Float16 v16h;
typedef __attribute__((ext_vector_type(8)))  _Float16 v8h;
typedef __attribute__((ext_vector_type(4)))  _Float16 v4h;
typedef __attribute__((ext_vector_type(8)))  float    v8f;

#define NODES_M 20000
#define NODES_G 80000
#define NTOT    100000
#define NREL    4
#define NBASE   10
#define DH      128

// ---------------- weight packing into WMMA B-fragment layout ----------------
// Pack layout: dst[((u*KT + t)*32 + lane)*16 + j]
//   n = u*16 + (lane&15); k = t*32 + ((lane<16)?0:8) + (j&7) + ((j>>3)<<4)   (B 32x16 fragment)
__global__ void pack_rel_w(const float* __restrict__ bases, const float* __restrict__ comp,
                           _Float16* __restrict__ dst)
{
  int idx = blockIdx.x * blockDim.x + threadIdx.x;  // 4*8*4*32*16 = 65536
  int j = idx & 15;
  int l = (idx >> 4) & 31;
  int t = (idx >> 9) & 3;
  int u = (idx >> 11) & 7;
  int r = idx >> 14;
  int k = t * 32 + ((l < 16) ? 0 : 8) + (j & 7) + ((j >> 3) << 4);
  int n = u * 16 + (l & 15);
  float v = 0.f;
#pragma unroll
  for (int b = 0; b < NBASE; ++b)
    v += comp[r * NBASE + b] * bases[((size_t)b * DH + k) * DH + n];
  dst[idx] = (_Float16)v;
}

__global__ void pack_plain_w(const float* __restrict__ W, _Float16* __restrict__ dst, int KT)
{
  int idx = blockIdx.x * blockDim.x + threadIdx.x;  // 8*KT*512 threads
  int j = idx & 15;
  int l = (idx >> 4) & 31;
  int ut = idx >> 9;
  int t = ut % KT;
  int u = ut / KT;
  if (u >= 8) return;
  int k = t * 32 + ((l < 16) ? 0 : 8) + (j & 7) + ((j >> 3) << 4);
  int n = u * 16 + (l & 15);
  dst[idx] = (_Float16)W[(size_t)k * DH + n];
}

// ---------------- projection GEMM: f32 A (on-the-fly cvt) x packed f16 B ----------------
// 8 waves/block; each WAVE owns a 16-row slab and the full 128-col output:
// one A-fragment load feeds 8 back-to-back WMMAs. B stays in global (L2-hot;
// LDS staging would cost 128-192KB and crush occupancy; proj is A-bandwidth bound).
__global__ void proj_wmma(const float* __restrict__ A, const _Float16* __restrict__ Bp,
                          const float* __restrict__ bias, _Float16* __restrict__ Xh,
                          int K, int rowOff, int Mslabs)
{
  const int lane = threadIdx.x & 31;
  const int slab = (blockIdx.x << 3) + (threadIdx.x >> 5);
  if (slab >= Mslabs) return;                  // wave-uniform: EXEC stays all-ones
  const int rowBase = slab << 4;
  const int KT = K >> 5;
  const int m  = lane & 15;
  const int kb = (lane < 16) ? 0 : 8;          // A-fragment K base (ISA 16-bit A 16x32)
  const float* arow = A + (size_t)(rowBase + m) * K + kb;

  v8f acc[8] = {};
  for (int t = 0; t < KT; ++t) {
    const float4* p4 = (const float4*)(arow + t * 32);   // 16B aligned
    float4 a0 = p4[0];   // K = kb+0..3
    float4 a1 = p4[1];   // K = kb+4..7
    float4 a2 = p4[4];   // K = 16+kb+0..3
    float4 a3 = p4[5];   // K = 16+kb+4..7
    v16h a;
    a[0]=(_Float16)a0.x; a[1]=(_Float16)a0.y; a[2]=(_Float16)a0.z; a[3]=(_Float16)a0.w;
    a[4]=(_Float16)a1.x; a[5]=(_Float16)a1.y; a[6]=(_Float16)a1.z; a[7]=(_Float16)a1.w;
    a[8]=(_Float16)a2.x; a[9]=(_Float16)a2.y; a[10]=(_Float16)a2.z; a[11]=(_Float16)a2.w;
    a[12]=(_Float16)a3.x; a[13]=(_Float16)a3.y; a[14]=(_Float16)a3.z; a[15]=(_Float16)a3.w;
#pragma unroll
    for (int u = 0; u < 8; ++u) {
      v16h b = *(const v16h*)(Bp + ((size_t)(u * KT + t) * 32 + lane) * 16);
      acc[u] = __builtin_amdgcn_wmma_f32_16x16x32_f16(false, a, false, b, (short)0, acc[u], false, false);
    }
  }
#pragma unroll
  for (int u = 0; u < 8; ++u) {
    const int col = u * 16 + m;
    const float bv = bias[col];
    _Float16* orow = Xh + (size_t)(rowOff + rowBase) * DH + col;
#pragma unroll
    for (int i = 0; i < 8; ++i) {
      int rr = (lane < 16) ? i : (i + 8);   // C/D layout: lanes 0-15 -> M=i, 16-31 -> M=i+8
      orow[(size_t)rr * DH] = (_Float16)(acc[u][i] + bv);
    }
  }
}

// ---------------- per-relation GEMM: f16 A (node feats) x packed f16 B ----------------
// grid.y = 5: r<4 -> Y[r] = Xh @ W_r ; r==4 -> Agg = Xh @ root + bias
// B (32KB for this relation) is staged in LDS once per block: the WMMA inner loop
// then runs off short-latency ds_load_b128 instead of per-WMMA L2 round trips.
__global__ void rel_wmma(const _Float16* __restrict__ Xh, const _Float16* __restrict__ Wp,
                         const _Float16* __restrict__ Rp, const float* __restrict__ bias,
                         float* __restrict__ Y, float* __restrict__ Agg, int Mslabs)
{
  __shared__ _Float16 Bs[4 * 8 * 32 * 16];     // 16384 halves = 32KB
  const int r = blockIdx.y;
  const _Float16* Bp = (r < NREL) ? (Wp + (size_t)r * 16384) : Rp;

  // cooperative stage: 32768B / (256 thr * 16B) = 8 iterations
  {
    const uint4* gs = (const uint4*)Bp;
    uint4* ls = (uint4*)Bs;
#pragma unroll
    for (int i = 0; i < 8; ++i) ls[threadIdx.x + i * 256] = gs[threadIdx.x + i * 256];
  }
  __syncthreads();

  const int lane = threadIdx.x & 31;
  const int slab = (blockIdx.x << 3) + (threadIdx.x >> 5);
  if (slab >= Mslabs) return;                  // wave-uniform
  const int rowBase = slab << 4;
  const int m  = lane & 15;
  const int kb = (lane < 16) ? 0 : 8;
  const _Float16* arow = Xh + (size_t)(rowBase + m) * DH + kb;

  v8f acc[8] = {};
#pragma unroll
  for (int t = 0; t < 4; ++t) {
    v8h lo = *(const v8h*)(arow + t * 32);        // K = kb..kb+7   (16B aligned)
    v8h hi = *(const v8h*)(arow + t * 32 + 16);   // K = 16+kb..16+kb+7
    v16h a = __builtin_shufflevector(lo, hi, 0,1,2,3,4,5,6,7,8,9,10,11,12,13,14,15);
#pragma unroll
    for (int u = 0; u < 8; ++u) {
      v16h b = *(const v16h*)(Bs + ((size_t)(u * 4 + t) * 32 + lane) * 16);
      acc[u] = __builtin_amdgcn_wmma_f32_16x16x32_f16(false, a, false, b, (short)0, acc[u], false, false);
    }
  }
  if (r < NREL) {
    float* obase = Y + ((size_t)r * NTOT + rowBase) * DH;
#pragma unroll
    for (int u = 0; u < 8; ++u) {
      const int col = u * 16 + m;
#pragma unroll
      for (int i = 0; i < 8; ++i) {
        int rr = (lane < 16) ? i : (i + 8);
        obase[(size_t)rr * DH + col] = acc[u][i];
      }
    }
  } else {
    float* obase = Agg + (size_t)rowBase * DH;
#pragma unroll
    for (int u = 0; u < 8; ++u) {
      const int col = u * 16 + m;
      const float bv = bias[col];
#pragma unroll
      for (int i = 0; i < 8; ++i) {
        int rr = (lane < 16) ? i : (i + 8);
        obase[(size_t)rr * DH + col] = acc[u][i] + bv;
      }
    }
  }
}

// ---------------- edge-side kernels ----------------
__global__ void zero_f32(float* __restrict__ p, int n)
{
  int i = blockIdx.x * blockDim.x + threadIdx.x;
  if (i < n) p[i] = 0.f;
}

__global__ void deg_count(const int* __restrict__ edst, const int* __restrict__ etyp,
                          float* __restrict__ deg, int E)
{
  int e = blockIdx.x * blockDim.x + threadIdx.x;
  if (e < E) unsafeAtomicAdd(&deg[(size_t)edst[e] * NREL + etyp[e]], 1.0f);
}

// 32 lanes per edge; edge index forced onto the scalar path (readfirstlane) so
// index/degree fetches become scalar loads; each lane does a float4 gather +
// 4 native global_atomic_add_f32.
__global__ void edge_scatter(const float* __restrict__ Y, const float* __restrict__ deg,
                             const int* __restrict__ esrc, const int* __restrict__ edst,
                             const int* __restrict__ etyp, float* __restrict__ Agg, int E)
{
  int e = __builtin_amdgcn_readfirstlane((blockIdx.x << 3) + (threadIdx.x >> 5));
  if (e >= E) return;
  int lane = threadIdx.x & 31;
  int s = __builtin_amdgcn_readfirstlane(esrc[e]);
  int d = __builtin_amdgcn_readfirstlane(edst[e]);
  int t = __builtin_amdgcn_readfirstlane(etyp[e]);
  float norm = 1.0f / fmaxf(deg[(size_t)d * NREL + t], 1.0f);
  float4 m4 = ((const float4*)(Y + ((size_t)t * NTOT + s) * DH))[lane];
  float* ap = Agg + (size_t)d * DH + lane * 4;
  unsafeAtomicAdd(ap + 0, m4.x * norm);
  unsafeAtomicAdd(ap + 1, m4.y * norm);
  unsafeAtomicAdd(ap + 2, m4.z * norm);
  unsafeAtomicAdd(ap + 3, m4.w * norm);
}

__global__ void f32_to_f16(const float* __restrict__ in, _Float16* __restrict__ out, int n4)
{
  int i = blockIdx.x * blockDim.x + threadIdx.x;
  if (i < n4) {
    float4 v = ((const float4*)in)[i];
    v4h h = { (_Float16)v.x, (_Float16)v.y, (_Float16)v.z, (_Float16)v.w };
    ((v4h*)out)[i] = h;
  }
}

// wave-per-label: 32 lanes cooperatively read one 512B row pair (fully coalesced),
// then wave32 shuffle reduction.
__global__ void score_kernel(const float* __restrict__ Out, const int* __restrict__ lsrc,
                             const int* __restrict__ ldst, float* __restrict__ sOut, int L)
{
  int l = __builtin_amdgcn_readfirstlane((blockIdx.x << 3) + (threadIdx.x >> 5));
  if (l >= L) return;
  int lane = threadIdx.x & 31;
  int ia = __builtin_amdgcn_readfirstlane(lsrc[l]);
  int ib = __builtin_amdgcn_readfirstlane(ldst[l]);
  float4 a = ((const float4*)(Out + (size_t)ia * DH))[lane];
  float4 b = ((const float4*)(Out + (size_t)(NODES_M + ib) * DH))[lane];
  float acc = a.x * b.x + a.y * b.y + a.z * b.z + a.w * b.w;
#pragma unroll
  for (int o = 16; o > 0; o >>= 1) acc += __shfl_xor(acc, o, 32);
  if (lane == 0) sOut[l] = acc;
}

// ---------------- host launch ----------------
extern "C" void kernel_launch(void* const* d_in, const int* in_sizes, int n_in,
                              void* d_out, int out_size, void* d_ws, size_t ws_size,
                              hipStream_t stream)
{
  const float* sim_m  = (const float*)d_in[0];
  const float* sim_g  = (const float*)d_in[1];
  const int*   esrc   = (const int*)d_in[2];
  const int*   edst   = (const int*)d_in[3];
  const int*   etyp   = (const int*)d_in[4];
  const int*   lsrc   = (const int*)d_in[5];
  const int*   ldst   = (const int*)d_in[6];
  const float* linmW  = (const float*)d_in[7];
  const float* linmB  = (const float*)d_in[8];
  const float* lingW  = (const float*)d_in[9];
  const float* lingB  = (const float*)d_in[10];
  const float* bases1 = (const float*)d_in[11];
  const float* comp1  = (const float*)d_in[12];
  const float* root1  = (const float*)d_in[13];
  const float* bias1  = (const float*)d_in[14];
  const float* bases2 = (const float*)d_in[15];
  const float* comp2  = (const float*)d_in[16];
  const float* root2  = (const float*)d_in[17];
  const float* bias2  = (const float*)d_in[18];
  const int E = in_sizes[2];
  const int L = in_sizes[5];

  char* ws = (char*)d_ws;
  size_t off = 0;
  auto alloc = [&](size_t bytes) -> void* {
    void* p = ws + off;
    off = (off + bytes + 255) & ~(size_t)255;
    return p;
  };
  _Float16* W1p = (_Float16*)alloc((size_t)4 * 16384 * 2);
  _Float16* W2p = (_Float16*)alloc((size_t)4 * 16384 * 2);
  _Float16* R1p = (_Float16*)alloc((size_t)16384 * 2);
  _Float16* R2p = (_Float16*)alloc((size_t)16384 * 2);
  _Float16* LMp = (_Float16*)alloc((size_t)8 * 16 * 512 * 2);
  _Float16* LGp = (_Float16*)alloc((size_t)8 * 24 * 512 * 2);
  float*    deg = (float*)alloc((size_t)NTOT * NREL * 4);
  _Float16* xh  = (_Float16*)alloc((size_t)NTOT * DH * 2);
  _Float16* hh  = (_Float16*)alloc((size_t)NTOT * DH * 2);
  float*    Y   = (float*)alloc((size_t)NREL * NTOT * DH * 4);
  float*    H   = (float*)alloc((size_t)NTOT * DH * 4);
  float*    O   = (float*)alloc((size_t)NTOT * DH * 4);

  // one-time weight packing (tiny)
  pack_rel_w<<<256, 256, 0, stream>>>(bases1, comp1, W1p);
  pack_rel_w<<<256, 256, 0, stream>>>(bases2, comp2, W2p);
  pack_plain_w<<<64, 256, 0, stream>>>(root1, R1p, 4);
  pack_plain_w<<<64, 256, 0, stream>>>(root2, R2p, 4);
  pack_plain_w<<<256, 256, 0, stream>>>(linmW, LMp, 16);
  pack_plain_w<<<384, 256, 0, stream>>>(lingW, LGp, 24);

  // input projections -> xh (f16): one wave per 16-row slab
  {
    int slabsM = NODES_M / 16;                 // 1250
    int slabsG = NODES_G / 16;                 // 5000
    proj_wmma<<<(slabsM + 7) / 8, 256, 0, stream>>>(sim_m, LMp, linmB, xh, 512, 0, slabsM);
    proj_wmma<<<(slabsG + 7) / 8, 256, 0, stream>>>(sim_g, LGp, lingB, xh, 768, NODES_M, slabsG);
  }

  // degree table (edge-set invariant; shared by both layers)
  zero_f32<<<(NTOT * NREL + 255) / 256, 256, 0, stream>>>(deg, NTOT * NREL);
  deg_count<<<(E + 255) / 256, 256, 0, stream>>>(edst, etyp, deg, E);

  const int slabsN = NTOT / 16;                // 6250
  const int relGx  = (slabsN + 7) / 8;

  // layer 1: per-relation transforms + root/bias init, then edge scatter
  rel_wmma<<<dim3(relGx, 5), 256, 0, stream>>>(xh, W1p, R1p, bias1, Y, H, slabsN);
  edge_scatter<<<(E + 7) / 8, 256, 0, stream>>>(Y, deg, esrc, edst, etyp, H, E);
  f32_to_f16<<<(NTOT * DH / 4 + 255) / 256, 256, 0, stream>>>(H, hh, NTOT * DH / 4);

  // layer 2 (reuse Y and deg)
  rel_wmma<<<dim3(relGx, 5), 256, 0, stream>>>(hh, W2p, R2p, bias2, Y, O, slabsN);
  edge_scatter<<<(E + 7) / 8, 256, 0, stream>>>(Y, deg, esrc, edst, etyp, O, E);

  // label dot products (wave per label)
  score_kernel<<<(L + 7) / 8, 256, 0, stream>>>(O, lsrc, ldst, (float*)d_out, L);

  (void)n_in; (void)out_size; (void)ws_size;
}
